// PSTTModule_42880953483414
// MI455X (gfx1250) — compile-verified, matrix operand order checked by reading, not run on hardware
//
#include <hip/hip_runtime.h>
#include <math.h>

// ---------------------------------------------------------------------------
// Problem constants (reference: B=8, N=1024, D=1152, H=8, K=10, M=20, S=1023)
// ---------------------------------------------------------------------------
#define C_B   8
#define C_N   1024
#define C_D   1152
#define C_H   8
#define C_DH  144
#define C_K   10
#define C_M   20
#define C_S   1023
#define C_3D  3456
#define C_2D  2304

typedef __attribute__((ext_vector_type(16))) __bf16 v16bf;
typedef __attribute__((ext_vector_type(8)))  float  v8f;

struct __align__(16) U128 { unsigned int x[4]; };
union FragBF { unsigned short s[16]; U128 q[2]; v16bf v; };

// fp32 -> bf16, round-to-nearest (ties away): add 0x8000, take high 16 bits.
__device__ __forceinline__ unsigned int f2bf_bits(float f) {
  union { float f; unsigned u; } cv; cv.f = f;
  return (cv.u + 0x8000u) >> 16;
}
__device__ __forceinline__ unsigned short f2bf(float f) {
  return (unsigned short)f2bf_bits(f);
}
// pack two floats -> two bf16 in one dword: 2x v_add + 1x v_perm_b32
__device__ __forceinline__ unsigned int pack_bf16_pair(float x, float y) {
  union { float f; unsigned u; } a, b; a.f = x; b.f = y;
  unsigned ua = a.u + 0x8000u;
  unsigned ub = b.u + 0x8000u;
  // bytes: dst = { ub[3], ub[2], ua[3], ua[2] }  -> { bf(y), bf(x) }
  return __builtin_amdgcn_perm(ub, ua, 0x07060302u);
}
// pack float4 -> 4 bf16 (8 bytes)
__device__ __forceinline__ uint2 f2bf4(float4 v) {
  uint2 r;
  r.x = pack_bf16_pair(v.x, v.y);
  r.y = pack_bf16_pair(v.z, v.w);
  return r;
}

// Load a 16-element bf16 fragment (A or B operand) from an LDS row whose
// K-values are contiguous. khalf = 0 (lanes 0-15) or 8 (lanes 16-31):
// elements 0..7  <- row[khalf + 0..7], elements 8..15 <- row[khalf + 16..23].
__device__ __forceinline__ void ld_frag(FragBF& f, const unsigned short* rowbase, int khalf) {
  f.q[0] = *reinterpret_cast<const U128*>(rowbase + khalf);
  f.q[1] = *reinterpret_cast<const U128*>(rowbase + khalf + 16);
}

#define WMMA_BF16(A_, B_, C_) \
  __builtin_amdgcn_wmma_f32_16x16x32_bf16(false, (A_), false, (B_), (short)0, (C_), false, false)

// ---------------------------------------------------------------------------
// Tiled GEMM:  C[M,N] = act(A[M,K] @ W[N,K]^T + bias)
// Block = 256 threads (8 waves), tile 128x64, K-chunks of 32 (Kdim % 32 == 0).
// Wave w: rows rt*32 (rt = w>>1), cols ct*32 (ct = w&1); 2x2 WMMA tiles.
// Out-of-range rows/cols are index-clamped on load (1:1 row/col mapping means
// garbage only lands in results that the guarded epilogue never stores).
// ---------------------------------------------------------------------------
__global__ __launch_bounds__(256) void gemm_bias_kernel(
    const float* __restrict__ A, const float* __restrict__ W,
    const float* __restrict__ bias, float* __restrict__ C,
    int Mrows, int Ncols, int Kdim, int act)
{
  __shared__ __align__(16) unsigned short As[128][32];   // 8 KB
  __shared__ __align__(16) unsigned short Bs[64][32];    // 4 KB
  const int tid  = threadIdx.x;
  const int lane = tid & 31;
  const int w    = tid >> 5;
  const int rt = w >> 1, ct = w & 1;
  const int m0 = blockIdx.y * 128;
  const int n0 = blockIdx.x * 64;
  const int khalf = (lane < 16) ? 0 : 8;

  v8f zero = {0.f,0.f,0.f,0.f,0.f,0.f,0.f,0.f};
  v8f acc00 = zero, acc01 = zero, acc10 = zero, acc11 = zero;

  // clamped load coordinates (branchless)
  int ar = m0 + (tid >> 3); if (ar >= Mrows) ar = Mrows - 1;          // A rows pass 0
  int ar2 = m0 + 64 + (tid >> 3); if (ar2 >= Mrows) ar2 = Mrows - 1;  // A rows pass 1
  int wn = n0 + (tid >> 3); if (wn >= Ncols) wn = Ncols - 1;          // W rows pass 0
  int wn2 = n0 + 32 + (tid >> 3); if (wn2 >= Ncols) wn2 = Ncols - 1;  // W rows pass 1
  const int c4 = (tid & 7) * 4;

  for (int kk = 0; kk < Kdim; kk += 32) {
    // A tile 128x32 -> bf16 LDS (float4 loads, uint2 stores)
    {
      float4 v0 = *reinterpret_cast<const float4*>(A + (size_t)ar  * Kdim + kk + c4);
      float4 v1 = *reinterpret_cast<const float4*>(A + (size_t)ar2 * Kdim + kk + c4);
      *reinterpret_cast<uint2*>(&As[(tid >> 3)][c4])      = f2bf4(v0);
      *reinterpret_cast<uint2*>(&As[64 + (tid >> 3)][c4]) = f2bf4(v1);
    }
    // W tile 64x32 -> bf16 LDS
    {
      float4 v0 = *reinterpret_cast<const float4*>(W + (size_t)wn  * Kdim + kk + c4);
      float4 v1 = *reinterpret_cast<const float4*>(W + (size_t)wn2 * Kdim + kk + c4);
      *reinterpret_cast<uint2*>(&Bs[(tid >> 3)][c4])      = f2bf4(v0);
      *reinterpret_cast<uint2*>(&Bs[32 + (tid >> 3)][c4]) = f2bf4(v1);
    }
    // prefetch next A K-slice (global_prefetch_b8) while we compute
    if (kk + 32 < Kdim) {
      int pr = m0 + (tid >> 1); if (pr >= Mrows) pr = Mrows - 1;
      __builtin_prefetch(A + (size_t)pr * Kdim + kk + 32 + (tid & 1) * 16, 0, 1);
    }
    __syncthreads();

    FragBF fa0, fa1, fb0, fb1;
    ld_frag(fa0, &As[rt * 32      + (lane & 15)][0], khalf);
    ld_frag(fa1, &As[rt * 32 + 16 + (lane & 15)][0], khalf);
    ld_frag(fb0, &Bs[ct * 32      + (lane & 15)][0], khalf);
    ld_frag(fb1, &Bs[ct * 32 + 16 + (lane & 15)][0], khalf);
    acc00 = WMMA_BF16(fa0.v, fb0.v, acc00);
    acc01 = WMMA_BF16(fa0.v, fb1.v, acc01);
    acc10 = WMMA_BF16(fa1.v, fb0.v, acc10);
    acc11 = WMMA_BF16(fa1.v, fb1.v, acc11);
    __syncthreads();
  }

  // guarded epilogue
  #pragma unroll
  for (int i = 0; i < 2; ++i) {
    int rbase = m0 + rt * 32 + i * 16 + ((lane < 16) ? 0 : 8);
    #pragma unroll
    for (int jc = 0; jc < 2; ++jc) {
      int col = n0 + ct * 32 + jc * 16 + (lane & 15);
      if (col >= Ncols) continue;
      float bv = bias[col];
      const v8f& a = (i == 0) ? (jc == 0 ? acc00 : acc01)
                              : (jc == 0 ? acc10 : acc11);
      #pragma unroll
      for (int j = 0; j < 8; ++j) {
        int row = rbase + j;
        if (row < Mrows) {
          float v = a[j] + bv;
          if (act == 1) v = 0.5f * v * (1.0f + erff(v * 0.70710678f));  // exact GELU
          C[(size_t)row * Ncols + col] = v;
        }
      }
    }
  }
}

// ---------------------------------------------------------------------------
// Fused flash attention over packed qkv [B, L, 3D] (per-token [q|k|v], head h
// = cols h*144..h*144+143 of each section). out[B,L,D] = softmax(QK^T/12) V.
// Block = 256 threads (8 waves); each wave owns 16 queries; key blocks of 32;
// dh = 144 padded to 160 (5 K-chunks of 32, pad zeroed once).
// ---------------------------------------------------------------------------
__global__ __launch_bounds__(256) void flash_attn_kernel(
    const float* __restrict__ qkv, float* __restrict__ out, int L)
{
  __shared__ __align__(16) unsigned short Ks[32][160];    // keys   [key][d]
  __shared__ __align__(16) unsigned short Vt[160][32];    // values [d][key]
  __shared__ __align__(16) unsigned short Pst[8][16][32]; // per-wave P staging
  const int tid  = threadIdx.x;
  const int lane = tid & 31;
  const int w    = tid >> 5;
  const int bh   = blockIdx.y;
  const int b = bh >> 3, h = bh & 7;
  const int q0 = blockIdx.x * 128 + w * 16;
  const int khalf = (lane < 16) ? 0 : 8;
  const float scale = 0.0833333333f;                      // 1/sqrt(144)

  // zero the d = 144..159 pad once (never rewritten)
  for (int i = tid; i < 512; i += 256) {
    Ks[i >> 4][144 + (i & 15)] = 0;
    Vt[144 + (i >> 5)][i & 31] = 0;
  }

  // Q fragments: row-clamped (clamped rows only affect unstored outputs)
  FragBF qf[5];
  {
    int r = q0 + (lane & 15); if (r >= L) r = L - 1;
    const float* qrow = qkv + ((size_t)b * L + r) * C_3D + h * C_DH;
    #pragma unroll
    for (int ch = 0; ch < 5; ++ch) {
      float4 lo0 = *reinterpret_cast<const float4*>(qrow + ch * 32 + khalf);
      float4 lo1 = *reinterpret_cast<const float4*>(qrow + ch * 32 + khalf + 4);
      uint2 p0 = f2bf4(lo0), p1 = f2bf4(lo1);
      qf[ch].q[0].x[0] = p0.x; qf[ch].q[0].x[1] = p0.y;
      qf[ch].q[0].x[2] = p1.x; qf[ch].q[0].x[3] = p1.y;
      if (ch < 4) {   // upper K-half: d = ch*32 + khalf + 16..23 (< 144 iff ch<4)
        float4 hi0 = *reinterpret_cast<const float4*>(qrow + ch * 32 + khalf + 16);
        float4 hi1 = *reinterpret_cast<const float4*>(qrow + ch * 32 + khalf + 20);
        uint2 q0p = f2bf4(hi0), q1p = f2bf4(hi1);
        qf[ch].q[1].x[0] = q0p.x; qf[ch].q[1].x[1] = q0p.y;
        qf[ch].q[1].x[2] = q1p.x; qf[ch].q[1].x[3] = q1p.y;
      } else {
        qf[ch].q[1].x[0] = 0; qf[ch].q[1].x[1] = 0;
        qf[ch].q[1].x[2] = 0; qf[ch].q[1].x[3] = 0;
      }
    }
  }

  v8f zero = {0.f,0.f,0.f,0.f,0.f,0.f,0.f,0.f};
  v8f O[9];
  #pragma unroll
  for (int t = 0; t < 9; ++t) O[t] = zero;
  float mrow[8], lrow[8];
  #pragma unroll
  for (int j = 0; j < 8; ++j) { mrow[j] = -1e30f; lrow[j] = 0.f; }

  const int nkb = (L + 31) / 32;
  for (int kb = 0; kb < nkb; ++kb) {
    __syncthreads();          // also covers the one-time pad init
    // K tile + transposed V tile: 32 keys x 36 float4 = 1152 vector loads
    #pragma unroll
    for (int t = 0; t < 5; ++t) {
      int idx = tid + t * 256;
      if (idx < 1152) {
        int i  = idx / 36;
        int d  = (idx - i * 36) * 4;
        int kg = kb * 32 + i;
        bool ok = (kg < L); if (!ok) kg = L - 1;
        const float* base = qkv + ((size_t)b * L + kg) * C_3D + h * C_DH + d;
        float4 kv4 = *reinterpret_cast<const float4*>(base + C_D);
        float4 vv4 = *reinterpret_cast<const float4*>(base + C_2D);
        if (!ok) { kv4 = make_float4(0.f,0.f,0.f,0.f); vv4 = kv4; }
        *reinterpret_cast<uint2*>(&Ks[i][d]) = f2bf4(kv4);
        Vt[d][i]     = f2bf(vv4.x);
        Vt[d + 1][i] = f2bf(vv4.y);
        Vt[d + 2][i] = f2bf(vv4.z);
        Vt[d + 3][i] = f2bf(vv4.w);
      }
    }
    __syncthreads();

    // S = Q K^T (two 16x16 tiles over this 32-key block)
    v8f a0 = zero, a1 = zero;
    #pragma unroll
    for (int ch = 0; ch < 5; ++ch) {
      FragBF k0, k1;
      ld_frag(k0, &Ks[(lane & 15)][ch * 32], khalf);
      ld_frag(k1, &Ks[16 + (lane & 15)][ch * 32], khalf);
      a0 = WMMA_BF16(qf[ch].v, k0.v, a0);
      a1 = WMMA_BF16(qf[ch].v, k1.v, a1);
    }

    // online softmax; row j lives in lanes 0-15 (half 0) / 16-31 (half 1)
    int c0 = kb * 32 + (lane & 15);
    int c1 = c0 + 16;
    float p0[8], p1[8], alpha[8];
    #pragma unroll
    for (int j = 0; j < 8; ++j) {
      float s0 = a0[j] * scale; if (c0 >= L) s0 = -1e30f;
      float s1 = a1[j] * scale; if (c1 >= L) s1 = -1e30f;
      float rm = fmaxf(s0, s1);
      rm = fmaxf(rm, __shfl_xor(rm, 1, 32));
      rm = fmaxf(rm, __shfl_xor(rm, 2, 32));
      rm = fmaxf(rm, __shfl_xor(rm, 4, 32));
      rm = fmaxf(rm, __shfl_xor(rm, 8, 32));
      float mn = fmaxf(mrow[j], rm);
      float al = __expf(mrow[j] - mn);
      float e0 = __expf(s0 - mn);
      float e1 = __expf(s1 - mn);
      float rs = e0 + e1;
      rs += __shfl_xor(rs, 1, 32);
      rs += __shfl_xor(rs, 2, 32);
      rs += __shfl_xor(rs, 4, 32);
      rs += __shfl_xor(rs, 8, 32);
      lrow[j] = lrow[j] * al + rs;
      mrow[j] = mn;
      alpha[j] = al;
      p0[j] = e0; p1[j] = e1;
    }
    #pragma unroll
    for (int t = 0; t < 9; ++t)
      #pragma unroll
      for (int j = 0; j < 8; ++j) O[t][j] *= alpha[j];

    // C-layout P -> A-layout P via per-wave LDS staging
    {
      int rofs = (lane < 16) ? 0 : 8;
      int cc = lane & 15;
      #pragma unroll
      for (int j = 0; j < 8; ++j) {
        Pst[w][rofs + j][cc]      = f2bf(p0[j]);
        Pst[w][rofs + j][cc + 16] = f2bf(p1[j]);
      }
    }
    FragBF pf;
    ld_frag(pf, &Pst[w][(lane & 15)][0], khalf);

    // O += P V  (9 d-tiles of 16)
    #pragma unroll
    for (int t = 0; t < 9; ++t) {
      FragBF vf;
      ld_frag(vf, &Vt[t * 16 + (lane & 15)][0], khalf);
      O[t] = WMMA_BF16(pf.v, vf.v, O[t]);
    }
  }

  // normalize + store
  {
    int rofs = (lane < 16) ? 0 : 8;
    #pragma unroll
    for (int j = 0; j < 8; ++j) {
      int row = q0 + rofs + j;
      if (row >= L) continue;
      float inv = 1.f / lrow[j];
      #pragma unroll
      for (int t = 0; t < 9; ++t) {
        int d = t * 16 + (lane & 15);
        if (d < C_DH)
          out[((size_t)b * L + row) * C_D + h * C_DH + d] = O[t][j] * inv;
      }
    }
  }
}

// ---------------------------------------------------------------------------
// Row-wise GEMV: out[r] = dot(A[r,:], w) + b[0]. One wave per row (Kdim%4==0).
// ---------------------------------------------------------------------------
__global__ __launch_bounds__(256) void gemv_rows_kernel(
    const float* __restrict__ A, const float* __restrict__ wv,
    const float* __restrict__ bs, float* __restrict__ outv, int R, int Kdim)
{
  int lane = threadIdx.x & 31;
  int w    = threadIdx.x >> 5;
  int row  = blockIdx.x * 8 + w;
  if (row >= R) return;
  const float4* ar = reinterpret_cast<const float4*>(A + (size_t)row * Kdim);
  const float4* wr = reinterpret_cast<const float4*>(wv);
  int K4 = Kdim >> 2;
  float s = 0.f;
  for (int k = lane; k < K4; k += 32) {
    float4 a = ar[k], b = wr[k];
    s += a.x * b.x + a.y * b.y + a.z * b.z + a.w * b.w;
  }
  s += __shfl_xor(s, 1, 32);  s += __shfl_xor(s, 2, 32);
  s += __shfl_xor(s, 4, 32);  s += __shfl_xor(s, 8, 32);
  s += __shfl_xor(s, 16, 32);
  if (lane == 0) outv[row] = s + bs[0];
}

// ---------------------------------------------------------------------------
// Per-batch top-Kt selection (descending, first-index ties like lax.top_k)
// and row gather. (Straight-through p-term == 0 in the forward pass.)
// ---------------------------------------------------------------------------
__global__ __launch_bounds__(256) void topk_gather_kernel(
    const float* __restrict__ scores, const float* __restrict__ src,
    float* __restrict__ dst, int Lr, int Kt, int Ddim)
{
  __shared__ float sc[1024];
  __shared__ int   idx[32];
  int b = blockIdx.x;
  int tid = threadIdx.x;
  for (int i = tid; i < 1024; i += 256)
    sc[i] = (i < Lr) ? scores[(size_t)b * Lr + i] : -1e30f;
  __syncthreads();
  if (tid == 0) {
    for (int t = 0; t < Kt; ++t) {
      int best = 0; float bv = sc[0];
      for (int i = 1; i < Lr; ++i) if (sc[i] > bv) { bv = sc[i]; best = i; }
      idx[t] = best; sc[best] = -1e30f;
    }
  }
  __syncthreads();
  int D4 = Ddim >> 2;
  for (int t = 0; t < Kt; ++t) {
    const float4* s4 = reinterpret_cast<const float4*>(src + ((size_t)b * Lr + idx[t]) * Ddim);
    float4* d4 = reinterpret_cast<float4*>(dst + ((size_t)b * Kt + t) * Ddim);
    for (int d = tid; d < D4; d += 256) d4[d] = s4[d];
  }
}

// ---------------------------------------------------------------------------
// Powerset masked-softmax pooling + LayerNorm:
//   subset s (bitmask s+1) -> weights over K seeds -> G[b,s,:] -> LN.
// ---------------------------------------------------------------------------
__global__ __launch_bounds__(256) void pool_powerset_kernel(
    const float* __restrict__ rawW, const float* __restrict__ xtopk,
    const float* __restrict__ g, const float* __restrict__ be,
    float* __restrict__ G)
{
  __shared__ float wsh[16];
  __shared__ __align__(16) float buf[C_D];
  __shared__ float r1[256], r2[256];
  int s = blockIdx.x, b = blockIdx.y;
  int tid = threadIdx.x;
  if (tid == 0) {
    unsigned mask = (unsigned)s + 1u;
    float rw[C_K];
    for (int k = 0; k < C_K; ++k) rw[k] = rawW[b * C_K + k];
    float mx = -1e30f;
    for (int k = 0; k < C_K; ++k) if ((mask >> k) & 1u) mx = fmaxf(mx, rw[k]);
    float sum = 0.f;
    for (int k = 0; k < C_K; ++k) {
      float e = ((mask >> k) & 1u) ? __expf(rw[k] - mx) : 0.f;
      wsh[k] = e; sum += e;
    }
    float inv = 1.f / sum;
    for (int k = 0; k < C_K; ++k) wsh[k] *= inv;
  }
  __syncthreads();
  float s1 = 0.f, s2 = 0.f;
  for (int d = tid; d < (C_D >> 2); d += 256) {
    float4 acc = make_float4(0.f, 0.f, 0.f, 0.f);
    #pragma unroll
    for (int k = 0; k < C_K; ++k) {
      float4 xv = reinterpret_cast<const float4*>(xtopk + ((size_t)b * C_K + k) * C_D)[d];
      float wk = wsh[k];
      acc.x += wk * xv.x; acc.y += wk * xv.y; acc.z += wk * xv.z; acc.w += wk * xv.w;
    }
    reinterpret_cast<float4*>(buf)[d] = acc;
    s1 += acc.x + acc.y + acc.z + acc.w;
    s2 += acc.x * acc.x + acc.y * acc.y + acc.z * acc.z + acc.w * acc.w;
  }
  r1[tid] = s1; r2[tid] = s2;
  __syncthreads();
  for (int st = 128; st > 0; st >>= 1) {
    if (tid < st) { r1[tid] += r1[tid + st]; r2[tid] += r2[tid + st]; }
    __syncthreads();
  }
  float mean = r1[0] / (float)C_D;
  float var  = r2[0] / (float)C_D - mean * mean;
  float rstd = rsqrtf(var + 1e-5f);
  float4* gr = reinterpret_cast<float4*>(G + ((size_t)b * C_S + s) * C_D);
  for (int d = tid; d < (C_D >> 2); d += 256) {
    float4 v = reinterpret_cast<const float4*>(buf)[d];
    float4 gg = reinterpret_cast<const float4*>(g)[d];
    float4 bb = reinterpret_cast<const float4*>(be)[d];
    v.x = (v.x - mean) * rstd * gg.x + bb.x;
    v.y = (v.y - mean) * rstd * gg.y + bb.y;
    v.z = (v.z - mean) * rstd * gg.z + bb.z;
    v.w = (v.w - mean) * rstd * gg.w + bb.w;
    gr[d] = v;
  }
}

// ---------------------------------------------------------------------------
// y = LayerNorm(x + r) * g + be, one block per row (in-place on x is safe).
// ---------------------------------------------------------------------------
__global__ __launch_bounds__(256) void residual_ln_kernel(
    const float* __restrict__ xin, const float* __restrict__ rin,
    const float* __restrict__ g, const float* __restrict__ be,
    float* __restrict__ y, int R)
{
  int row = blockIdx.x;
  if (row >= R) return;
  __shared__ __align__(16) float buf[C_D];
  __shared__ float r1[256], r2[256];
  int tid = threadIdx.x;
  const float4* xr = reinterpret_cast<const float4*>(xin + (size_t)row * C_D);
  const float4* rr = reinterpret_cast<const float4*>(rin + (size_t)row * C_D);
  float s1 = 0.f, s2 = 0.f;
  for (int d = tid; d < (C_D >> 2); d += 256) {
    float4 a = xr[d], bq = rr[d];
    float4 v = make_float4(a.x + bq.x, a.y + bq.y, a.z + bq.z, a.w + bq.w);
    reinterpret_cast<float4*>(buf)[d] = v;
    s1 += v.x + v.y + v.z + v.w;
    s2 += v.x * v.x + v.y * v.y + v.z * v.z + v.w * v.w;
  }
  r1[tid] = s1; r2[tid] = s2;
  __syncthreads();
  for (int st = 128; st > 0; st >>= 1) {
    if (tid < st) { r1[tid] += r1[tid + st]; r2[tid] += r2[tid + st]; }
    __syncthreads();
  }
  float mean = r1[0] / (float)C_D;
  float var  = r2[0] / (float)C_D - mean * mean;
  float rstd = rsqrtf(var + 1e-5f);
  float4* yr = reinterpret_cast<float4*>(y + (size_t)row * C_D);
  for (int d = tid; d < (C_D >> 2); d += 256) {
    float4 v = reinterpret_cast<const float4*>(buf)[d];
    float4 gg = reinterpret_cast<const float4*>(g)[d];
    float4 bb = reinterpret_cast<const float4*>(be)[d];
    v.x = (v.x - mean) * rstd * gg.x + bb.x;
    v.y = (v.y - mean) * rstd * gg.y + bb.y;
    v.z = (v.z - mean) * rstd * gg.z + bb.z;
    v.w = (v.w - mean) * rstd * gg.w + bb.w;
    yr[d] = v;
  }
}

// ---------------------------------------------------------------------------
// Cross attention with M=20 keys: one thread per (b, h, query).
// kv rows: [b*M + m][0..1151]=k, [1152..2303]=v.
// ---------------------------------------------------------------------------
__global__ __launch_bounds__(256) void cross_attn_kernel(
    const float* __restrict__ q, const float* __restrict__ kv,
    float* __restrict__ ox)
{
  int gid = blockIdx.x * 256 + threadIdx.x;
  if (gid >= C_B * C_H * C_N) return;
  int n = gid & (C_N - 1);
  int h = (gid >> 10) & (C_H - 1);
  int b = gid >> 13;
  const float4* qr = reinterpret_cast<const float4*>(q + ((size_t)b * C_N + n) * C_D + h * C_DH);
  float sc[C_M];
  float mx = -1e30f;
  for (int m = 0; m < C_M; ++m) {
    const float4* kr = reinterpret_cast<const float4*>(kv + ((size_t)b * C_M + m) * C_2D + h * C_DH);
    float s = 0.f;
    for (int d = 0; d < (C_DH >> 2); ++d) {
      float4 a = qr[d], bq = kr[d];
      s += a.x * bq.x + a.y * bq.y + a.z * bq.z + a.w * bq.w;
    }
    s *= 0.0833333333f;
    sc[m] = s; mx = fmaxf(mx, s);
  }
  float sum = 0.f;
  for (int m = 0; m < C_M; ++m) { sc[m] = __expf(sc[m] - mx); sum += sc[m]; }
  float inv = 1.f / sum;
  float4* orow = reinterpret_cast<float4*>(ox + ((size_t)b * C_N + n) * C_D + h * C_DH);
  for (int d = 0; d < (C_DH >> 2); ++d) {
    float4 acc = make_float4(0.f, 0.f, 0.f, 0.f);
    for (int m = 0; m < C_M; ++m) {
      float4 v = reinterpret_cast<const float4*>(kv + ((size_t)b * C_M + m) * C_2D + C_D + h * C_DH)[d];
      float p = sc[m];
      acc.x += p * v.x; acc.y += p * v.y; acc.z += p * v.z; acc.w += p * v.w;
    }
    acc.x *= inv; acc.y *= inv; acc.z *= inv; acc.w *= inv;
    orow[d] = acc;
  }
}

// out = x + y * sigmoid(gate_logit)   (y lives in d_out; in-place)
__global__ __launch_bounds__(256) void final_gate_kernel(
    const float* __restrict__ x, const float* __restrict__ gl,
    float* __restrict__ out, int n4)
{
  int i = blockIdx.x * 256 + threadIdx.x;
  if (i >= n4) return;
  float gate = 1.f / (1.f + __expf(-gl[0]));
  float4 a = reinterpret_cast<const float4*>(x)[i];
  float4 yv = reinterpret_cast<float4*>(out)[i];
  yv.x = a.x + yv.x * gate; yv.y = a.y + yv.y * gate;
  yv.z = a.z + yv.z * gate; yv.w = a.w + yv.w * gate;
  reinterpret_cast<float4*>(out)[i] = yv;
}

// ---------------------------------------------------------------------------
extern "C" void kernel_launch(void* const* d_in, const int* in_sizes, int n_in,
                              void* d_out, int out_size, void* d_ws, size_t ws_size,
                              hipStream_t stream)
{
  (void)in_sizes; (void)n_in; (void)out_size; (void)ws_size;
  const float* x          = (const float*)d_in[0];
  const float* pair_w_in  = (const float*)d_in[1];
  const float* pair_b_in  = (const float*)d_in[2];
  const float* pair_w_out = (const float*)d_in[3];
  const float* pair_b_out = (const float*)d_in[4];
  const float* pair_ln_g  = (const float*)d_in[5];
  const float* pair_ln_b  = (const float*)d_in[6];
  const float* obj_w1     = (const float*)d_in[7];
  const float* obj_b1     = (const float*)d_in[8];
  const float* obj_w2     = (const float*)d_in[9];
  const float* obj_b2     = (const float*)d_in[10];
  const float* pool_w     = (const float*)d_in[11];
  const float* pool_b     = (const float*)d_in[12];
  const float* pool_ln_g  = (const float*)d_in[13];
  const float* pool_ln_b  = (const float*)d_in[14];
  const float* grp_w_in   = (const float*)d_in[15];
  const float* grp_b_in   = (const float*)d_in[16];
  const float* grp_w_out  = (const float*)d_in[17];
  const float* grp_b_out  = (const float*)d_in[18];
  const float* grp_ln_g   = (const float*)d_in[19];
  const float* grp_ln_b   = (const float*)d_in[20];
  const float* grps_w     = (const float*)d_in[21];
  const float* grps_b     = (const float*)d_in[22];
  const float* crs_w_in   = (const float*)d_in[23];
  const float* crs_b_in   = (const float*)d_in[24];
  const float* crs_w_out  = (const float*)d_in[25];
  const float* crs_b_out  = (const float*)d_in[26];
  const float* crs_ln_g   = (const float*)d_in[27];
  const float* crs_ln_b   = (const float*)d_in[28];
  const float* out_w1     = (const float*)d_in[29];
  const float* out_b1     = (const float*)d_in[30];
  const float* out_w2     = (const float*)d_in[31];
  const float* out_b2     = (const float*)d_in[32];
  const float* gate_logit = (const float*)d_in[33];

  const int BN = C_B * C_N;   // 8192 token rows
  const int BS = C_B * C_S;   // 8184 group rows

  float* ws = (float*)d_ws;
  size_t off = 0;
  float* qkvbuf = ws + off; off += (size_t)BN * C_3D;          // 113 MB
  float* bufA   = ws + off; off += (size_t)BN * C_D;           // 37.7 MB
  float* bufB   = ws + off; off += (size_t)BN * C_D;           // 37.7 MB
  float* scores = ws + off; off += (size_t)BN;
  float* gs     = ws + off; off += (size_t)BS + 8;
  float* xtopk  = ws + off; off += (size_t)C_B * C_K * C_D;
  float* rawW   = ws + off; off += (size_t)C_B * C_K + 8;
  float* Gr     = ws + off; off += (size_t)C_B * C_M * C_D;
  float* kvbuf  = ws + off; off += (size_t)C_B * C_M * C_2D;
  float* bufC   = (float*)d_out;                               // d_out as scratch

  dim3 blk(256);
  auto gg = [](int Nc, int Mr) { return dim3((Nc + 63) / 64, (Mr + 127) / 128); };

  // ---- Step 1: pair attention ----
  gemm_bias_kernel<<<gg(C_3D, BN), blk, 0, stream>>>(x, pair_w_in, pair_b_in, qkvbuf, BN, C_3D, C_D, 0);
  flash_attn_kernel<<<dim3(8, C_B * C_H), blk, 0, stream>>>(qkvbuf, bufA, C_N);
  gemm_bias_kernel<<<gg(C_D, BN), blk, 0, stream>>>(bufA, pair_w_out, pair_b_out, bufC, BN, C_D, C_D, 0);
  residual_ln_kernel<<<dim3(BN), blk, 0, stream>>>(x, bufC, pair_ln_g, pair_ln_b, bufB, BN); // ctx

  // objectness scores + top-k seed gather (from original x)
  gemm_bias_kernel<<<gg(288, BN), blk, 0, stream>>>(bufB, obj_w1, obj_b1, bufA, BN, 288, C_D, 1);
  gemv_rows_kernel<<<dim3((BN + 7) / 8), blk, 0, stream>>>(bufA, obj_w2, obj_b2, scores, BN, 288);
  topk_gather_kernel<<<dim3(C_B), blk, 0, stream>>>(scores, x, xtopk, C_N, C_K, C_D);

  // ---- Step 2: powerset pooling ----
  gemv_rows_kernel<<<dim3((C_B * C_K + 7) / 8), blk, 0, stream>>>(xtopk, pool_w, pool_b, rawW, C_B * C_K, C_D);
  pool_powerset_kernel<<<dim3(C_S, C_B), blk, 0, stream>>>(rawW, xtopk, pool_ln_g, pool_ln_b, bufB); // G

  // ---- Step 3: group self-attention + top-M ----
  gemm_bias_kernel<<<gg(C_3D, BS), blk, 0, stream>>>(bufB, grp_w_in, grp_b_in, qkvbuf, BS, C_3D, C_D, 0);
  flash_attn_kernel<<<dim3(8, C_B * C_H), blk, 0, stream>>>(qkvbuf, bufA, C_S);
  gemm_bias_kernel<<<gg(C_D, BS), blk, 0, stream>>>(bufA, grp_w_out, grp_b_out, bufC, BS, C_D, C_D, 0);
  residual_ln_kernel<<<dim3(BS), blk, 0, stream>>>(bufB, bufC, grp_ln_g, grp_ln_b, bufB, BS); // G final
  gemv_rows_kernel<<<dim3((BS + 7) / 8), blk, 0, stream>>>(bufB, grps_w, grps_b, gs, BS, C_D);
  topk_gather_kernel<<<dim3(C_B), blk, 0, stream>>>(gs, bufB, Gr, C_S, C_M, C_D);

  // ---- Step 4: cross attention + gated MLP residual ----
  gemm_bias_kernel<<<gg(C_D, BN), blk, 0, stream>>>(x, crs_w_in, crs_b_in, bufA, BN, C_D, C_D, 0);   // q
  gemm_bias_kernel<<<gg(C_2D, C_B * C_M), blk, 0, stream>>>(Gr, crs_w_in + (size_t)C_D * C_D,
                                                            crs_b_in + C_D, kvbuf, C_B * C_M, C_2D, C_D, 0);
  cross_attn_kernel<<<dim3((C_B * C_H * C_N + 255) / 256), blk, 0, stream>>>(bufA, kvbuf, bufC);
  gemm_bias_kernel<<<gg(C_D, BN), blk, 0, stream>>>(bufC, crs_w_out, crs_b_out, qkvbuf, BN, C_D, C_D, 0);
  residual_ln_kernel<<<dim3(BN), blk, 0, stream>>>(x, qkvbuf, crs_ln_g, crs_ln_b, bufB, BN);         // xr
  gemm_bias_kernel<<<gg(C_D, BN), blk, 0, stream>>>(bufB, out_w1, out_b1, bufA, BN, C_D, C_D, 1);
  gemm_bias_kernel<<<gg(C_D, BN), blk, 0, stream>>>(bufA, out_w2, out_b2, bufC, BN, C_D, C_D, 0);
  final_gate_kernel<<<dim3((BN * C_D / 4 + 255) / 256), blk, 0, stream>>>(x, gate_logit, bufC, BN * C_D / 4);
}
